// HyperDiffWave_4664334483988
// MI455X (gfx1250) — compile-verified, matrix operand order checked by reading, not run on hardware
//
#include <hip/hip_runtime.h>
#include <hip/hip_bf16.h>

// ---------------------------------------------------------------- types
typedef __attribute__((ext_vector_type(8)))  float  v8f;
typedef __attribute__((ext_vector_type(16))) __bf16 v16bf;
typedef __attribute__((ext_vector_type(4)))  unsigned int v4u_t;
typedef __attribute__((ext_vector_type(8)))  int v8i_t;
typedef __attribute__((ext_vector_type(4)))  int v4i_t;

union FragBF { v16bf v; uint4 q[2]; };

#define RES 64
#define LAYERS 40
#define LSEQ 16384
#define NB 8

static __device__ __forceinline__ unsigned short f2bf(float f) {
  unsigned int u = __float_as_uint(f);
  u = (u + 0x7FFFu + ((u >> 16) & 1u)) >> 16;     // round-to-nearest-even
  return (unsigned short)u;
}
static __device__ __forceinline__ float bf2f(unsigned short h) {
  return __uint_as_float(((unsigned int)h) << 16);
}
static __device__ __forceinline__ float lrelu(float x) { return x >= 0.f ? x : 0.2f * x; }

// gfx1250 native transcendentals: V_TANH_F32 / V_EXP_F32 / V_RCP_F32 are TRANS
// ops that co-execute with the WMMA pipe (ISA 05_wmma: XDL ops tracked as TRANS).
#if __has_builtin(__builtin_amdgcn_tanhf)
static __device__ __forceinline__ float fast_tanh(float x) { return __builtin_amdgcn_tanhf(x); }
#elif __has_builtin(__builtin_amdgcn_tanh_f32)
static __device__ __forceinline__ float fast_tanh(float x) { return __builtin_amdgcn_tanh_f32(x); }
#else
static __device__ __forceinline__ float fast_tanh(float x) { return tanhf(x); }
#endif

#if __has_builtin(__builtin_amdgcn_rcpf)
static __device__ __forceinline__ float fast_rcp(float x) { return __builtin_amdgcn_rcpf(x); }
#else
static __device__ __forceinline__ float fast_rcp(float x) { return 1.f / x; }
#endif

static __device__ __forceinline__ float sigm(float x) { return fast_rcp(1.f + __expf(-x)); }

// Load a 16x32 bf16 A/B fragment from LDS stored as [rowcol][64 K-halves].
// CDNA5 16-bit fragment layout: lanes 0-15 hold K {0..7,16..23}, lanes 16-31
// hold K {8..15,24..31} of K-block kb, for row/col = (lane&15) of the tile.
static __device__ __forceinline__ FragBF ldfrag(const unsigned short* base, int rc, int kb, int hi) {
  FragBF f;
  const unsigned short* p = base + rc * 64 + kb * 32 + hi * 8;
  f.q[0] = *(const uint4*)(p);
  f.q[1] = *(const uint4*)(p + 16);
  return f;
}

// ---------------------------------------------------------------- diffusion embedding
__global__ void emb_kernel(const float* tstep, const float* We1, const float* be1,
                           const float* We2, const float* be2, float* eout) {
  const int n = blockIdx.x, t = threadIdx.x;      // 512 threads
  __shared__ float feats[128];
  __shared__ float e1[512];
  if (t < 64) {
    float freq = __powf(10.f, (float)t * (4.f / 63.f));
    float arg = tstep[n] * freq;
    feats[t] = __sinf(arg);
    feats[t + 64] = __cosf(arg);
  }
  __syncthreads();
  const float* wr = We1 + (size_t)t * 128;
  float a = be1[t];
  for (int c = 0; c < 128; ++c) a += wr[c] * feats[c];
  e1[t] = a * sigm(a);
  __syncthreads();
  const float* wr2 = We2 + (size_t)t * 512;
  a = be2[t];
  for (int c = 0; c < 512; ++c) a += wr2[c] * e1[c];
  eout[(size_t)n * 512 + t] = a * sigm(a);
}

// ---------------------------------------------------------------- predictor stage 1 (per-group MLP)
__global__ void pred_stage1(const float* e, int Cin, int G,
                            const float* ws, const float* bs,
                            const float* w1, const float* b1,
                            const float* w2, const float* b2, float* H2) {
  const int g = blockIdx.x, n = blockIdx.y, t = threadIdx.x;   // 64 threads
  __shared__ float hs[64];
  __shared__ float rs[64];
  const float* ev = e + (size_t)n * Cin;
  const float* wr = ws + (size_t)(g * 64 + t) * Cin;
  float a = bs[g * 64 + t];
  for (int c = 0; c < Cin; ++c) a += wr[c] * ev[c];
  float h = lrelu(a);
  hs[t] = h;
  __syncthreads();
  const float* w1r = w1 + ((size_t)g * 64 + t) * 64;
  a = b1[g * 64 + t];
  for (int c = 0; c < 64; ++c) a += w1r[c] * hs[c];
  rs[t] = lrelu(a);
  __syncthreads();
  const float* w2r = w2 + ((size_t)g * 64 + t) * 64;
  a = b2[g * 64 + t];
  for (int c = 0; c < 64; ++c) a += w2r[c] * rs[c];
  H2[((size_t)n * G + g) * 64 + t] = h + lrelu(a);
}

// ---------------------------------------------------------------- predictor stage 2 (bandwidth-bound GEMVs)
// np: emit bf16 conv weights permuted to [layer][tap][o][c] for direct LDS use.
__global__ void np_stage2(const float* H2, const float* we, const float* be,
                          unsigned short* dw) {
  __shared__ float h2s[8][64];
  const int g = blockIdx.y;
  const int m = blockIdx.x * 256 + threadIdx.x;           // 0..24575  (= k*8192 + o*64 + c)
  for (int i = threadIdx.x; i < 512; i += 256)
    h2s[i >> 6][i & 63] = H2[((size_t)(i >> 6) * LAYERS + g) * 64 + (i & 63)];
  __syncthreads();
  const int k = m >> 13;
  const int oc = m & 8191;
  const int j = oc * 3 + k;                               // source index in [o][c][k] order
  const float* wr = we + ((size_t)g * 24576 + j) * 64;
  float b = be[(size_t)g * 24576 + j];
  float acc[8];
#pragma unroll
  for (int s = 0; s < 8; ++s) acc[s] = b;
  for (int c = 0; c < 64; ++c) {
    float wv = wr[c];
#pragma unroll
    for (int s = 0; s < 8; ++s) acc[s] += wv * h2s[s][c];
  }
#pragma unroll
  for (int s = 0; s < 8; ++s)
    dw[((size_t)s * LAYERS + g) * 24576 + m] = f2bf(acc[s]);
}

__global__ void sp_stage2(const float* H2, const float* we, const float* be,
                          unsigned short* W2) {
  __shared__ float h2s[8][64];
  const int g = blockIdx.y;                               // 0..38
  const int m = blockIdx.x * 256 + threadIdx.x;           // 0..8191
  for (int i = threadIdx.x; i < 512; i += 256)
    h2s[i >> 6][i & 63] = H2[((size_t)(i >> 6) * (LAYERS - 1) + g) * 64 + (i & 63)];
  __syncthreads();
  const float* wr = we + ((size_t)g * 8192 + m) * 64;
  float b = be[(size_t)g * 8192 + m];
  float acc[8];
#pragma unroll
  for (int s = 0; s < 8; ++s) acc[s] = b;
  for (int c = 0; c < 64; ++c) {
    float wv = wr[c];
#pragma unroll
    for (int s = 0; s < 8; ++s) acc[s] += wv * h2s[s][c];
  }
#pragma unroll
  for (int s = 0; s < 8; ++s)
    W2[((size_t)s * LAYERS + g) * 8192 + m] = f2bf(acc[s]);
}

// Final-layer weights: rows 0..63 zero (no x update), rows 64..127 = fw (skip).
__global__ void fp_stage2(const float* H2, const float* we, const float* be,
                          unsigned short* W2) {
  __shared__ float h2s[8][64];
  const int m = blockIdx.x * 256 + threadIdx.x;           // 0..8191
  for (int i = threadIdx.x; i < 512; i += 256)
    h2s[i >> 6][i & 63] = H2[(size_t)(i >> 6) * 64 + (i & 63)];
  __syncthreads();
  if (m < 4096) {
#pragma unroll
    for (int s = 0; s < 8; ++s)
      W2[((size_t)s * LAYERS + (LAYERS - 1)) * 8192 + m] = 0;
  } else {
    const int j = m - 4096;
    const float* wr = we + (size_t)j * 64;
    float b = be[j];
    float acc[8];
#pragma unroll
    for (int s = 0; s < 8; ++s) acc[s] = b;
    for (int c = 0; c < 64; ++c) {
      float wv = wr[c];
#pragma unroll
      for (int s = 0; s < 8; ++s) acc[s] += wv * h2s[s][c];
    }
#pragma unroll
    for (int s = 0; s < 8; ++s)
      W2[((size_t)s * LAYERS + (LAYERS - 1)) * 8192 + m] = f2bf(acc[s]);
  }
}

// ---------------------------------------------------------------- input projection + skip clear
__global__ void inproj(const float* audio, const float* W_in, const float* b_in,
                       unsigned short* x0, float* skip) {
  size_t i = (size_t)blockIdx.x * 256 + threadIdx.x;      // over 8*64*L
  const int l = (int)(i % LSEQ);
  size_t t = i / LSEQ;
  const int c = (int)(t % RES);
  const int n = (int)(t / RES);
  float v = audio[(size_t)n * LSEQ + l] * W_in[c] + b_in[c];
  x0[i] = f2bf(v > 0.f ? v : 0.f);
  skip[i] = 0.f;
}

// ---------------------------------------------------------------- the 40x fused WMMA layer
struct __align__(16) LayerSmem {
  unsigned short wA[3 * 128 * 64];   // conv weights [tap][o][c]   48KB  (TDM target, offset 0)
  unsigned short win[3 * 128 * 64];  // x windows    [tap][col][c] 48KB
  unsigned short w2[128 * 64];       // 1x1 weights  [o][c]        16KB
  unsigned short zb[128 * 64];       // gated z      [col][c]      16KB
};
static_assert(sizeof(LayerSmem) == 131072, "LDS layout");

__global__ __launch_bounds__(256)
void layer_kernel(const unsigned short* __restrict__ x_in,
                  unsigned short* __restrict__ x_out,
                  float* __restrict__ skip,
                  const unsigned short* __restrict__ dw_all,
                  const unsigned short* __restrict__ w2_all,
                  int layer, int d) {
  __shared__ LayerSmem s;
  const int tid = threadIdx.x;
  const int n = blockIdx.y;
  const int l0 = blockIdx.x * 128;

  // --- TDM: async DMA this (sample,layer)'s 48KB conv-weight block into LDS (offset 0)
  const unsigned short* dwp = dw_all + ((size_t)n * LAYERS + layer) * 24576;
  if (tid < 32) {  // one wave issues the descriptor
    unsigned long long ga = (unsigned long long)(const void*)dwp;
    v4u_t g0 = { 1u,                                     // count=1, user mode
                 0u,                                     // lds_addr: wA is at LDS offset 0
                 (unsigned int)ga,                       // global_addr[31:0]
                 ((unsigned int)(ga >> 32) & 0x01FFFFFFu) | 0x80000000u }; // addr[56:32] | type=2
    v8i_t g1 = { 0x00010000,                             // data_size = 2 bytes
                 0x60000000,                             // tensor_dim0 = 24576 (low16 in [63:48])
                 0x00010000,                             // tensor_dim1 = 1
                 0x60000000,                             // tile_dim0 = 24576
                 1,                                      // tile_dim1 = 1
                 24576,                                  // tensor_dim0_stride
                 0x60000000,                             // tensor_dim1_stride = 24576
                 0 };
    v4i_t g2 = { 0, 0, 0, 0 };
    v4i_t g3 = { 0, 0, 0, 0 };
    v8i_t g4 = { 0, 0, 0, 0, 0, 0, 0, 0 };               // clang-23 form: extra group, zero-filled
    __builtin_amdgcn_tensor_load_to_lds(g0, g1, g2, g3, g4, 0);
  }

  // --- cooperative loads overlapped with the TDM transfer
  { // 1x1 (speaker / final) weights: 16KB contiguous bf16
    const uint4* src = (const uint4*)(w2_all + ((size_t)n * LAYERS + layer) * 8192);
    uint4* dst = (uint4*)s.w2;
    for (int i = tid; i < 1024; i += 256) dst[i] = src[i];
  }
  // three dilated x windows, zero-padded at sequence edges, stored [tap][col][c]
  for (int idx = tid; idx < 3 * 64 * 128; idx += 256) {
    const int col = idx & 127;
    const int c = (idx >> 7) & 63;
    const int k = idx >> 13;
    const int g = l0 + col + (k - 1) * d;
    unsigned short v = 0;
    if (g >= 0 && g < LSEQ) v = x_in[((size_t)n * RES + c) * LSEQ + g];
    s.win[k * 8192 + col * 64 + c] = v;
  }
  if (tid < 32) __builtin_amdgcn_s_wait_tensorcnt(0);
  __syncthreads();

  const int lane = tid & 31;
  const int lo16 = lane & 15;
  const int hi = lane >> 4;
  const int w = tid >> 5;       // wave 0..7
  const int m0 = w & 3;         // M-tile pair: rows 16*m0.. and 64+16*m0..
  const int nhalf = w >> 2;     // column half

  // --- GEMM1: y = dconv(x)  (M=128, K=3*64, N=128), gate tanh*sigmoid in-registers
  for (int nt = 0; nt < 4; ++nt) {
    const int ntile = nhalf * 4 + nt;
    v8f accA = { 0.f, 0.f, 0.f, 0.f, 0.f, 0.f, 0.f, 0.f };  // rows  0..63  (tanh arg)
    v8f accB = { 0.f, 0.f, 0.f, 0.f, 0.f, 0.f, 0.f, 0.f };  // rows 64..127 (sigmoid arg)
#pragma unroll
    for (int k = 0; k < 3; ++k) {
#pragma unroll
      for (int kb = 0; kb < 2; ++kb) {
        FragBF b  = ldfrag(s.win + k * 8192, ntile * 16 + lo16, kb, hi);
        FragBF aA = ldfrag(s.wA + k * 8192, m0 * 16 + lo16, kb, hi);
        FragBF aB = ldfrag(s.wA + k * 8192, (m0 + 4) * 16 + lo16, kb, hi);
        accA = __builtin_amdgcn_wmma_f32_16x16x32_bf16(false, aA.v, false, b.v, (short)0, accA, false, false);
        accB = __builtin_amdgcn_wmma_f32_16x16x32_bf16(false, aB.v, false, b.v, (short)0, accB, false, false);
      }
    }
#pragma unroll
    for (int r = 0; r < 8; ++r) {
      const int row = m0 * 16 + r + hi * 8;     // z channel 0..63
      const int col = ntile * 16 + lo16;
      float z = fast_tanh(accA[r]) * sigm(accB[r]);   // v_tanh_f32 / v_exp_f32 / v_rcp_f32
      s.zb[col * 64 + row] = f2bf(z);
    }
  }
  __syncthreads();

  // --- GEMM2: y2 = W2(128x64) @ z(64x128); rows<64 -> x update, rows>=64 -> skip
  for (int nt = 0; nt < 4; ++nt) {
    const int ntile = nhalf * 4 + nt;
    v8f accA = { 0.f, 0.f, 0.f, 0.f, 0.f, 0.f, 0.f, 0.f };
    v8f accB = { 0.f, 0.f, 0.f, 0.f, 0.f, 0.f, 0.f, 0.f };
#pragma unroll
    for (int kb = 0; kb < 2; ++kb) {
      FragBF b  = ldfrag(s.zb, ntile * 16 + lo16, kb, hi);
      FragBF aA = ldfrag(s.w2, m0 * 16 + lo16, kb, hi);
      FragBF aB = ldfrag(s.w2, (m0 + 4) * 16 + lo16, kb, hi);
      accA = __builtin_amdgcn_wmma_f32_16x16x32_bf16(false, aA.v, false, b.v, (short)0, accA, false, false);
      accB = __builtin_amdgcn_wmma_f32_16x16x32_bf16(false, aB.v, false, b.v, (short)0, accB, false, false);
    }
#pragma unroll
    for (int r = 0; r < 8; ++r) {
      const int row = m0 * 16 + r + hi * 8;     // channel 0..63
      const int col = ntile * 16 + lo16;
      const size_t gi = ((size_t)n * RES + row) * LSEQ + (l0 + col);
      float xc = bf2f(s.win[8192 + col * 64 + row]);        // center window = x
      x_out[gi] = f2bf((xc + accA[r]) * 0.70710678f);
      skip[gi] += accB[r];                                   // fp32 skip accumulation
    }
  }
}

// ---------------------------------------------------------------- output projection
__global__ void outproj(const float* skip, const float* Wo1, const float* bo1,
                        const float* Wo2, const float* bo2, float* out) {
  __shared__ float st[64 * 128];
  __shared__ float w1s[64 * 64];
  __shared__ float part[2][128];
  const int n = blockIdx.y;
  const int l0 = blockIdx.x * 128;
  const int tid = threadIdx.x;    // 256
  for (int i = tid; i < 4096; i += 256) w1s[i] = Wo1[i];
  const float scale = 0.15811388f;  // 1/sqrt(40)
  for (int i = tid; i < 8192; i += 256) {
    const int col = i & 127, c = i >> 7;
    st[c * 128 + col] = skip[((size_t)n * RES + c) * LSEQ + l0 + col] * scale;
  }
  __syncthreads();
  const int col = tid & 127, half = tid >> 7;
  float acc = 0.f;
  for (int o = half * 32; o < half * 32 + 32; ++o) {
    float a = bo1[o];
    for (int c = 0; c < 64; ++c) a += w1s[o * 64 + c] * st[c * 128 + col];
    acc += Wo2[o] * (a > 0.f ? a : 0.f);
  }
  part[half][col] = acc;
  __syncthreads();
  if (tid < 128) out[(size_t)n * LSEQ + l0 + tid] = part[0][tid] + part[1][tid] + bo2[0];
}

// ---------------------------------------------------------------- host driver
extern "C" void kernel_launch(void* const* d_in, const int* in_sizes, int n_in,
                              void* d_out, int out_size, void* d_ws, size_t ws_size,
                              hipStream_t stream) {
  (void)in_sizes; (void)n_in; (void)out_size; (void)ws_size;
  const float* audio = (const float*)d_in[0];
  const float* dstep = (const float*)d_in[1];
  const float* spk   = (const float*)d_in[2];
  const float* W_in  = (const float*)d_in[3];
  const float* b_in  = (const float*)d_in[4];
  const float* Wo1   = (const float*)d_in[5];
  const float* bo1   = (const float*)d_in[6];
  const float* Wo2   = (const float*)d_in[7];
  const float* bo2   = (const float*)d_in[8];
  const float* We1   = (const float*)d_in[9];
  const float* be1   = (const float*)d_in[10];
  const float* We2   = (const float*)d_in[11];
  const float* be2   = (const float*)d_in[12];
  const float* np_ws = (const float*)d_in[13];
  const float* np_bs = (const float*)d_in[14];
  const float* np_w1 = (const float*)d_in[15];
  const float* np_b1 = (const float*)d_in[16];
  const float* np_w2 = (const float*)d_in[17];
  const float* np_b2 = (const float*)d_in[18];
  const float* np_we = (const float*)d_in[19];
  const float* np_be = (const float*)d_in[20];
  const float* sp_ws = (const float*)d_in[21];
  const float* sp_bs = (const float*)d_in[22];
  const float* sp_w1 = (const float*)d_in[23];
  const float* sp_b1 = (const float*)d_in[24];
  const float* sp_w2 = (const float*)d_in[25];
  const float* sp_b2 = (const float*)d_in[26];
  const float* sp_we = (const float*)d_in[27];
  const float* sp_be = (const float*)d_in[28];
  const float* fp_ws = (const float*)d_in[29];
  const float* fp_bs = (const float*)d_in[30];
  const float* fp_w1 = (const float*)d_in[31];
  const float* fp_b1 = (const float*)d_in[32];
  const float* fp_w2 = (const float*)d_in[33];
  const float* fp_b2 = (const float*)d_in[34];
  const float* fp_we = (const float*)d_in[35];
  const float* fp_be = (const float*)d_in[36];

  char* ws = (char*)d_ws;
  size_t off = 0;
  auto take = [&](size_t b) { size_t o = off; off += (b + 255) & ~(size_t)255; return o; };
  float* e_emb = (float*)(ws + take((size_t)NB * 512 * 4));
  float* H2np  = (float*)(ws + take((size_t)NB * LAYERS * 64 * 4));
  float* H2sp  = (float*)(ws + take((size_t)NB * (LAYERS - 1) * 64 * 4));
  float* H2fp  = (float*)(ws + take((size_t)NB * 64 * 4));
  unsigned short* dw_bf = (unsigned short*)(ws + take((size_t)NB * LAYERS * 24576 * 2));
  unsigned short* W2_bf = (unsigned short*)(ws + take((size_t)NB * LAYERS * 8192 * 2));
  unsigned short* xA = (unsigned short*)(ws + take((size_t)NB * RES * LSEQ * 2));
  unsigned short* xB = (unsigned short*)(ws + take((size_t)NB * RES * LSEQ * 2));
  float* skip = (float*)(ws + take((size_t)NB * RES * LSEQ * 4));

  emb_kernel<<<NB, 512, 0, stream>>>(dstep, We1, be1, We2, be2, e_emb);
  pred_stage1<<<dim3(LAYERS, NB), 64, 0, stream>>>(e_emb, 512, LAYERS, np_ws, np_bs, np_w1, np_b1, np_w2, np_b2, H2np);
  pred_stage1<<<dim3(LAYERS - 1, NB), 64, 0, stream>>>(spk, 256, LAYERS - 1, sp_ws, sp_bs, sp_w1, sp_b1, sp_w2, sp_b2, H2sp);
  pred_stage1<<<dim3(1, NB), 64, 0, stream>>>(spk, 256, 1, fp_ws, fp_bs, fp_w1, fp_b1, fp_w2, fp_b2, H2fp);
  np_stage2<<<dim3(96, LAYERS), 256, 0, stream>>>(H2np, np_we, np_be, dw_bf);
  sp_stage2<<<dim3(32, LAYERS - 1), 256, 0, stream>>>(H2sp, sp_we, sp_be, W2_bf);
  fp_stage2<<<dim3(32, 1), 256, 0, stream>>>(H2fp, fp_we, fp_be, W2_bf);
  inproj<<<(NB * RES * LSEQ) / 256, 256, 0, stream>>>(audio, W_in, b_in, xA, skip);

  static const int DTAB[8] = { 1, 3, 9, 27, 81, 243, 729, 2187 };
  unsigned short* xin = xA;
  unsigned short* xout = xB;
  for (int i = 0; i < LAYERS; ++i) {
    layer_kernel<<<dim3(LSEQ / 128, NB), 256, 0, stream>>>(xin, xout, skip, dw_bf, W2_bf, i, DTAB[i & 7]);
    unsigned short* t = xin; xin = xout; xout = t;
  }
  outproj<<<dim3(LSEQ / 128, NB), 256, 0, stream>>>(skip, Wo1, bo1, Wo2, bo2, (float*)d_out);
}
